// TGCNLongTerm_60765197304391
// MI455X (gfx1250) — matrix operand order repaired and checked by reading, not compiled, not run on hardware
//
#include <hip/hip_runtime.h>
#include <hip/hip_bf16.h>
#include <stdint.h>

#define NN 4096   // nodes
#define WW 12     // window
#define HH 256    // hidden
#define PP 10     // horizon

typedef __attribute__((ext_vector_type(16))) __bf16 v16bf;
typedef __attribute__((ext_vector_type(8)))  __bf16 bh8;
typedef __attribute__((ext_vector_type(8)))  float  v8f;

struct Halves { bh8 lo; bh8 hi; };
union TileU { v16bf v; Halves p; };

__device__ __forceinline__ __bf16 f2bf(float f) {
  unsigned u = __builtin_bit_cast(unsigned, f);
  unsigned r = (u + 0x7FFFu + ((u >> 16) & 1u)) >> 16;
  return __builtin_bit_cast(__bf16, (unsigned short)r);
}

// ---- WMMA tile loaders -----------------------------------------------------
// A (16x32 bf16, row-major, ld elems): lane half h, row m=lane&15.
// contiguous K = 16*half .. 16*half+15 (matches B loaders)
__device__ __forceinline__ v16bf load_a_tile(const __bf16* A, int lda, int m, int half) {
  TileU u;
  const __bf16* p0 = A + (long)m * lda + 16 * half;
  u.p.lo = *(const bh8*)p0;
  u.p.hi = *(const bh8*)(p0 + 8);
  return u.v;
}
// B (32x16) supplied as Bt (N x K row-major): lane = column n, contiguous
// K = 16*half .. 16*half+15
__device__ __forceinline__ v16bf load_b_tile(const __bf16* Bt, int ldb, int n, int half) {
  TileU u;
  const __bf16* p0 = Bt + (long)n * ldb + 16 * half;
  u.p.lo = *(const bh8*)p0;
  u.p.hi = *(const bh8*)(p0 + 8);
  return u.v;
}

// ---- Simple per-wave WMMA GEMM (used for skinny L @ window, Ntiles==1) -----
__global__ void gemm_bf16_wmma(const __bf16* __restrict__ A, int lda,
                               const __bf16* __restrict__ Bt, int ldb,
                               float* __restrict__ Cf, int ldc,
                               int Mtiles, int Ntiles, int K) {
  int wave = (int)((blockIdx.x * blockDim.x + threadIdx.x) >> 5);
  int lane = threadIdx.x & 31;
  if (wave >= Mtiles * Ntiles) return;     // wave-uniform: EXEC stays all-ones
  int tm = wave / Ntiles, tn = wave % Ntiles;
  int r = lane & 15, half = lane >> 4;

  const __bf16* Ab = A + (size_t)(tm * 16) * lda;
  const __bf16* Bb = Bt + (size_t)(tn * 16) * ldb;

  v8f acc;
  #pragma unroll
  for (int i = 0; i < 8; ++i) acc[i] = 0.0f;

  for (int k = 0; k < K; k += 32) {
    v16bf a = load_a_tile(Ab + k, lda, r, half);
    v16bf b = load_b_tile(Bb + k, ldb, r, half);
    if (k + 64 < K)
      __builtin_prefetch((const void*)(Ab + (long)r * lda + k + 64), 0, 3);
    acc = __builtin_amdgcn_wmma_f32_16x16x32_bf16(
        false, a, false, b, (short)0, acc, false, false);
  }

  int n = tn * 16 + r;
  #pragma unroll
  for (int i = 0; i < 8; ++i)
    Cf[(size_t)(tm * 16 + i + 8 * half) * ldc + n] = acc[i];
}

// ---- Register-blocked WMMA GEMM with async-LDS B-panel staging -------------
// Block = 256 threads = 8 waves. Block tile: 256 (M) x 64 (N).
// Wave w owns rows [blk*256 + w*32, +32) = 2 M-tiles x 4 N-tiles = 8 WMMAs/k.
// B panel (64 cols x 32 K, bf16) is shared by all 8 waves: streamed into LDS
// with global_load_async_to_lds_b128 (ASYNCcnt), double-buffered with partial
// waits (s_wait_asynccnt 1), consumed via ds_load_b128 (80B padded rows).
#define BROW 40   // LDS row stride in bf16 elems (80B: pad 64B->80B vs banks)
__global__ void __launch_bounds__(256)
gemm_bf16_wmma_blk(const __bf16* __restrict__ A, int lda,
                   const __bf16* __restrict__ Bt, int ldb,
                   float* __restrict__ Cf, __bf16* __restrict__ Cb, int ldc,
                   const float* __restrict__ bias, int relu, int K) {
  __shared__ __bf16 Blds[2][64 * BROW];

  int tid  = threadIdx.x;
  int wv   = tid >> 5, lane = tid & 31;
  int r    = lane & 15, half = lane >> 4;
  size_t mBase = (size_t)blockIdx.y * 256 + (size_t)wv * 32;
  int    nBase = blockIdx.x * 64;

  int colc  = tid >> 2;        // 0..63 : which B column this thread copies
  int chunk = tid & 3;         // 0..3  : 16-byte chunk within the 64B row
  const __bf16* srcBase = Bt + (size_t)(nBase + colc) * ldb + chunk * 8;

  v8f acc[2][4];
  #pragma unroll
  for (int i = 0; i < 2; ++i)
    #pragma unroll
    for (int j = 0; j < 4; ++j)
      #pragma unroll
      for (int t = 0; t < 8; ++t) acc[i][j][t] = 0.0f;

  const int nk = K >> 5;

  // prologue: stage k-chunk 0 into buffer 0
  {
    unsigned dst = (unsigned)(size_t)&Blds[0][colc * BROW + chunk * 8];
    asm volatile("global_load_async_to_lds_b128 %0, %1, off"
                 :: "v"(dst), "v"(srcBase) : "memory");
  }

  for (int kk = 0; kk < nk; ++kk) {
    int cur = kk & 1;
    if (kk + 1 < nk) {
      // stream next panel while current one is consumed
      unsigned dst = (unsigned)(size_t)&Blds[1 - cur][colc * BROW + chunk * 8];
      const __bf16* src = srcBase + (kk + 1) * 32;
      asm volatile("global_load_async_to_lds_b128 %0, %1, off"
                   :: "v"(dst), "v"(src) : "memory");
      asm volatile("s_wait_asynccnt 1" ::: "memory");  // oldest (cur) done
    } else {
      asm volatile("s_wait_asynccnt 0" ::: "memory");
    }
    __syncthreads();                                    // cur panel visible

    // A fragments: 2 M-tiles, global b128 loads in WMMA layout
    v16bf afr[2];
    #pragma unroll
    for (int i = 0; i < 2; ++i)
      afr[i] = load_a_tile(A + (mBase + i * 16) * lda + kk * 32, lda, r, half);

    // B fragments: 4 N-tiles from LDS
    v16bf bfr[4];
    #pragma unroll
    for (int j = 0; j < 4; ++j) {
      const __bf16* bp = &Blds[cur][(j * 16 + r) * BROW + half * 16];
      TileU u;
      u.p.lo = *(const bh8*)bp;
      u.p.hi = *(const bh8*)(bp + 8);
      bfr[j] = u.v;
    }

    #pragma unroll
    for (int i = 0; i < 2; ++i)
      #pragma unroll
      for (int j = 0; j < 4; ++j)
        acc[i][j] = __builtin_amdgcn_wmma_f32_16x16x32_bf16(
            false, afr[i], false, bfr[j], (short)0, acc[i][j], false, false);

    __syncthreads();   // all waves done reading cur before it is re-staged
  }

  #pragma unroll
  for (int i = 0; i < 2; ++i) {
    #pragma unroll
    for (int j = 0; j < 4; ++j) {
      int n = nBase + j * 16 + r;
      float bv = bias ? bias[n] : 0.0f;
      #pragma unroll
      for (int t = 0; t < 8; ++t) {
        size_t m = mBase + i * 16 + t + 8 * half;
        float v = acc[i][j][t] + bv;
        if (relu) v = v > 0.0f ? v : 0.0f;
        if (Cf) Cf[m * ldc + n] = v;
        if (Cb) Cb[m * ldc + n] = f2bf(v);
      }
    }
  }
}

// ---- Setup / elementwise kernels ------------------------------------------
__global__ void fill_u32_k(unsigned* p, unsigned v, long n) {
  long i = (long)blockIdx.x * blockDim.x + threadIdx.x;
  if (i < n) p[i] = v;
}
__global__ void fill_f32_k(float* p, float v, long n) {
  long i = (long)blockIdx.x * blockDim.x + threadIdx.x;
  if (i < n) p[i] = v;
}
__global__ void deg_accum_k(const int* __restrict__ ei, const float* __restrict__ ew,
                            float* deg, int E) {
  int e = blockIdx.x * blockDim.x + threadIdx.x;
  if (e < E) atomicAdd(&deg[ei[e]], ew[e]);
}
__global__ void rsqrt_k(const float* deg, float* dv, int n) {
  int i = blockIdx.x * blockDim.x + threadIdx.x;
  if (i < n) dv[i] = rsqrtf(deg[i]);       // deg >= 1 (self loop), never inf
}
// L[i][j] = d[i] * adj[j][i] * d[j]; edge e: adj[src][dst]=w  =>  L[dst][src]
__global__ void scatter_L_k(const int* __restrict__ ei, const float* __restrict__ ew,
                            const float* __restrict__ dv, __bf16* L, int E) {
  int e = blockIdx.x * blockDim.x + threadIdx.x;
  if (e >= E) return;
  int s = ei[e], t = ei[E + e];
  L[(size_t)t * NN + s] = f2bf(dv[t] * ew[e] * dv[s]);
}
__global__ void diag_L_k(const float* __restrict__ dv, __bf16* L, int n) {
  int i = blockIdx.x * blockDim.x + threadIdx.x;
  if (i < n) L[(size_t)i * NN + i] = f2bf(dv[i] * dv[i]);
}
// taT[j][k] = ta_w[k][j]   (ta_w is (W*H, H) row-major)
__global__ void conv_taT_k(const float* __restrict__ ta_w, __bf16* taT) {
  long i = (long)blockIdx.x * blockDim.x + threadIdx.x;
  if (i >= (long)HH * (WW * HH)) return;
  int j = (int)(i / (WW * HH)), k = (int)(i % (WW * HH));
  taT[i] = f2bf(ta_w[(size_t)k * HH + j]);
}
__global__ void conv_bf16_k(const float* __restrict__ src, __bf16* dst, long n) {
  long i = (long)blockIdx.x * blockDim.x + threadIdx.x;
  if (i < n) dst[i] = f2bf(src[i]);
}
// WbT[w][node] = x[node][w]  (rows 12..15 zero padding)
__global__ void init_win_k(const float* __restrict__ x, __bf16* WbT) {
  long i = (long)blockIdx.x * blockDim.x + threadIdx.x;
  if (i >= (long)16 * NN) return;
  int w = (int)(i / NN), n = (int)(i % NN);
  WbT[i] = (w < WW) ? f2bf(x[(size_t)n * WW + w]) : f2bf(0.0f);
}
// G[n][w*H+h] = tanh(ax[n][w] * gcn_w[h])
__global__ void build_g_k(const float* __restrict__ ax, const float* __restrict__ gcn_w,
                          __bf16* __restrict__ G) {
  long i = (long)blockIdx.x * blockDim.x + threadIdx.x;
  if (i >= (long)NN * WW * HH) return;
  int n = (int)(i / (WW * HH));
  int c = (int)(i % (WW * HH));
  int w = c >> 8, h = c & 255;
  G[i] = f2bf(tanhf(ax[n * 16 + w] * gcn_w[h]));
}
// GRU gate combine + h update + prediction dot + window shift
__global__ void gru_k(const float* __restrict__ gx, const float* __restrict__ gh,
                      const float* __restrict__ b_ih, const float* __restrict__ b_hh,
                      float* __restrict__ h, __bf16* __restrict__ hb,
                      const float* __restrict__ out_w, const float* __restrict__ out_b,
                      __bf16* __restrict__ WbT, float* __restrict__ out, int p) {
  int n = blockIdx.x, j = threadIdx.x;
  const float* gxr = gx + (size_t)n * (3 * HH);
  const float* ghr = gh + (size_t)n * (3 * HH);
  float xr = gxr[j] + b_ih[j];
  float xz = gxr[j + HH] + b_ih[j + HH];
  float xn = gxr[j + 2 * HH] + b_ih[j + 2 * HH];
  float hr = ghr[j] + b_hh[j];
  float hz = ghr[j + HH] + b_hh[j + HH];
  float hn = ghr[j + 2 * HH] + b_hh[j + 2 * HH];
  float rr = 1.0f / (1.0f + __expf(-(xr + hr)));
  float zz = 1.0f / (1.0f + __expf(-(xz + hz)));
  float nn = tanhf(xn + rr * hn);
  float hold = h[(size_t)n * HH + j];
  float hnew = (1.0f - zz) * nn + zz * hold;
  h[(size_t)n * HH + j] = hnew;
  hb[(size_t)n * HH + j] = f2bf(hnew);

  __shared__ float red[HH];
  red[j] = hnew * out_w[j];
  __syncthreads();
  for (int s = HH / 2; s > 0; s >>= 1) {
    if (j < s) red[j] += red[j + s];
    __syncthreads();
  }
  if (j == 0) {
    float pred = red[0] + out_b[0];
    out[(size_t)n * PP + p] = pred;
    for (int w = 0; w < WW - 1; ++w) WbT[(size_t)w * NN + n] = WbT[(size_t)(w + 1) * NN + n];
    WbT[(size_t)(WW - 1) * NN + n] = f2bf(pred);
  }
}

// ---- Host ------------------------------------------------------------------
extern "C" void kernel_launch(void* const* d_in, const int* in_sizes, int n_in,
                              void* d_out, int out_size, void* d_ws, size_t ws_size,
                              hipStream_t stream) {
  const float* x     = (const float*)d_in[0];
  const int*   ei    = (const int*)d_in[1];
  const float* ew    = (const float*)d_in[2];
  const float* gcn_w = (const float*)d_in[3];
  const float* ta_w  = (const float*)d_in[4];
  const float* ta_b  = (const float*)d_in[5];
  const float* w_ih  = (const float*)d_in[6];
  const float* w_hh  = (const float*)d_in[7];
  const float* b_ih  = (const float*)d_in[8];
  const float* b_hh  = (const float*)d_in[9];
  const float* out_w = (const float*)d_in[10];
  const float* out_b = (const float*)d_in[11];
  float* out = (float*)d_out;
  int E = in_sizes[2];

  char* w8 = (char*)d_ws;
  size_t off = 0;
  auto alloc = [&](size_t bytes) -> void* {
    void* p = w8 + off;
    off += (bytes + 255) & ~(size_t)255;
    return p;
  };
  __bf16* L   = (__bf16*)alloc((size_t)NN * NN * 2);          // 33.5 MB (L2-resident)
  __bf16* G   = (__bf16*)alloc((size_t)NN * WW * HH * 2);     // 25 MB
  float*  gx  = (float*) alloc((size_t)NN * 3 * HH * 4);      // 12.6 MB
  float*  gh  = (float*) alloc((size_t)NN * 3 * HH * 4);      // 12.6 MB
  __bf16* tb  = (__bf16*)alloc((size_t)NN * HH * 2);
  __bf16* hb  = (__bf16*)alloc((size_t)NN * HH * 2);
  float*  h   = (float*) alloc((size_t)NN * HH * 4);
  __bf16* WbT = (__bf16*)alloc((size_t)16 * NN * 2);
  float*  ax  = (float*) alloc((size_t)NN * 16 * 4);
  __bf16* taT = (__bf16*)alloc((size_t)HH * WW * HH * 2);
  __bf16* wih = (__bf16*)alloc((size_t)3 * HH * HH * 2);
  __bf16* whh = (__bf16*)alloc((size_t)3 * HH * HH * 2);
  float*  deg = (float*) alloc((size_t)NN * 4);
  float*  dv  = (float*) alloc((size_t)NN * 4);

  auto blocks = [](long n) { return dim3((unsigned)((n + 255) / 256)); };

  // ---- setup (re-done every call: deterministic, graph-capture safe) -------
  fill_u32_k<<<blocks((long)NN * NN / 2), 256, 0, stream>>>((unsigned*)L, 0u, (long)NN * NN / 2);
  fill_f32_k<<<blocks(NN), 256, 0, stream>>>(deg, 1.0f, NN);            // self-loop degree
  fill_f32_k<<<blocks((long)NN * HH), 256, 0, stream>>>(h, 0.0f, (long)NN * HH);
  fill_u32_k<<<blocks((long)NN * HH / 2), 256, 0, stream>>>((unsigned*)hb, 0u, (long)NN * HH / 2);
  deg_accum_k<<<blocks(E), 256, 0, stream>>>(ei, ew, deg, E);
  rsqrt_k<<<blocks(NN), 256, 0, stream>>>(deg, dv, NN);
  scatter_L_k<<<blocks(E), 256, 0, stream>>>(ei, ew, dv, L, E);
  diag_L_k<<<blocks(NN), 256, 0, stream>>>(dv, L, NN);
  conv_taT_k<<<blocks((long)HH * WW * HH), 256, 0, stream>>>(ta_w, taT);
  conv_bf16_k<<<blocks((long)3 * HH * HH), 256, 0, stream>>>(w_ih, wih, (long)3 * HH * HH);
  conv_bf16_k<<<blocks((long)3 * HH * HH), 256, 0, stream>>>(w_hh, whh, (long)3 * HH * HH);
  init_win_k<<<blocks((long)16 * NN), 256, 0, stream>>>(x, WbT);

  // ---- P sequential horizon steps -----------------------------------------
  for (int p = 0; p < PP; ++p) {
    // ax(4096x16) = L(4096x4096) @ window(4096x16)  — skinny, L2-resident
    gemm_bf16_wmma<<<dim3((NN / 16 * 32 + 255) / 256), 256, 0, stream>>>(
        L, NN, WbT, NN, ax, 16, NN / 16, 1, NN);
    // G = tanh(ax outer gcn_w)
    build_g_k<<<blocks((long)NN * WW * HH), 256, 0, stream>>>(ax, gcn_w, G);
    // t = relu(G @ ta_w + b) -> bf16       M=4096 K=3072 N=256
    gemm_bf16_wmma_blk<<<dim3(HH / 64, NN / 256), 256, 0, stream>>>(
        G, WW * HH, taT, WW * HH, nullptr, tb, HH, ta_b, 1, WW * HH);
    // gx = t @ w_ih^T                      M=4096 K=256 N=768
    gemm_bf16_wmma_blk<<<dim3(3 * HH / 64, NN / 256), 256, 0, stream>>>(
        tb, HH, wih, HH, gx, nullptr, 3 * HH, nullptr, 0, HH);
    // gh = h @ w_hh^T
    gemm_bf16_wmma_blk<<<dim3(3 * HH / 64, NN / 256), 256, 0, stream>>>(
        hb, HH, whh, HH, gh, nullptr, 3 * HH, nullptr, 0, HH);
    // gates + h update + prediction + window shift
    gru_k<<<NN, HH, 0, stream>>>(gx, gh, b_ih, b_hh, h, hb, out_w, out_b, WbT, out, p);
  }
}
